// MoE_23622320128375
// MI455X (gfx1250) — compile-verified, matrix operand order checked by reading, not run on hardware
//
#include <hip/hip_runtime.h>
#include <hip/hip_bf16.h>

// Problem constants (from reference): B=2, S=2048 -> T=4096, H=1024, I=2048, E=8, K=2
#define T_TOK 4096
#define H_DIM 1024
#define I_DIM 2048
#define E_NUM 8
#define K_TOP 2

typedef __bf16 bf16_t;
typedef bf16_t v16bf __attribute__((ext_vector_type(16)));
typedef float  v8f   __attribute__((ext_vector_type(8)));

// ---------------- helpers ----------------

__device__ __forceinline__ unsigned short f2bf_raw(float f) {
  union { float f; unsigned u; } un; un.f = f;
  unsigned r = un.u + 0x7FFFu + ((un.u >> 16) & 1u);   // round-to-nearest-even
  return (unsigned short)(r >> 16);
}

__device__ __forceinline__ bf16_t f2bf(float f) {
  unsigned short h = f2bf_raw(f);
  union { unsigned short s; bf16_t b; } o; o.s = h; return o.b;
}

// Async global->LDS copy, 16B per lane. VDST operand carries the per-lane LDS
// byte offset (low 32 bits of a generic shared-memory address), VADDR the
// 64-bit global address. Tracked by ASYNCcnt.
__device__ __forceinline__ void async_load_b128(unsigned lds_off, const void* gaddr) {
  asm volatile("global_load_async_to_lds_b128 %0, %1, off"
               :: "v"(lds_off), "v"((unsigned long long)(size_t)gaddr)
               : "memory");
}

template <int N>
__device__ __forceinline__ void wait_asynccnt() {
  asm volatile("s_wait_asynccnt %0" :: "i"(N) : "memory");
}

// Load one 16x32 bf16 WMMA operand fragment from an LDS tile stored row-major
// with `stride` bf16 elements per row. Per CDNA5 layout: lane l (0..15) holds
// row (l&15), K chunks {0..7,16..23}; lane l+16 holds K chunks {8..15,24..31}.
__device__ __forceinline__ v16bf load_frag(const bf16_t* rowbase, int stride, int lane) {
  const bf16_t* p = rowbase + (lane & 15) * stride + ((lane >> 4) << 3);
  union { uint4 u[2]; v16bf v; } un;
  un.u[0] = *(const uint4*)(p);        // K 0..7   (or 8..15)
  un.u[1] = *(const uint4*)(p + 16);   // K 16..23 (or 24..31)
  return un.v;
}

// ---------------- kernels ----------------

__global__ void moe_init(int* __restrict__ cnt) {
  if (threadIdx.x < E_NUM) cnt[threadIdx.x] = 0;
}

// fp32 activations -> bf16 (one pass)
__global__ __launch_bounds__(256) void moe_cvt_x(const float4* __restrict__ x,
                                                 uint2* __restrict__ xbf) {
  int i = blockIdx.x * 256 + threadIdx.x;     // exactly T*H/4 threads
  float4 v = x[i];
  uint2 o;
  o.x = (unsigned)f2bf_raw(v.x) | ((unsigned)f2bf_raw(v.y) << 16);
  o.y = (unsigned)f2bf_raw(v.z) | ((unsigned)f2bf_raw(v.w) << 16);
  xbf[i] = o;
}

// Transpose + convert one expert weight matrix: in fp32 [K][N] -> out bf16 [N][K].
// Tile 64(K) x 32(N); both global read and write fully coalesced via LDS bounce.
__global__ __launch_bounds__(256) void moe_transpose_cvt(
    const float* __restrict__ in, bf16_t* __restrict__ out, int Kdim, int Ndim)
{
  __shared__ bf16_t L[32 * 74];                 // [n][k], pad 74 -> conflict-free
  const int e  = blockIdx.z;
  const int k0 = blockIdx.x * 64;
  const int n0 = blockIdx.y * 32;
  const float* src = in + (size_t)e * Kdim * Ndim;
  bf16_t*      dst = out + (size_t)e * Kdim * Ndim;

  const int tid = threadIdx.x;
  {
    const int col = tid & 31;                   // n within tile
    const int rb  = tid >> 5;                   // base k row
#pragma unroll
    for (int i = 0; i < 8; ++i) {
      int k = rb + i * 8;                       // 0..63
      L[col * 74 + k] = f2bf(src[(size_t)(k0 + k) * Ndim + (n0 + col)]);
    }
  }
  __syncthreads();
  {
    const int c  = tid & 31;                    // u32 chunk within k-row (0..31)
    const int nb = tid >> 5;                    // base n
#pragma unroll
    for (int i = 0; i < 4; ++i) {
      int n = nb + i * 8;                       // 0..31
      unsigned v = *(const unsigned*)&L[n * 74 + c * 2];
      *(unsigned*)(dst + (size_t)(n0 + n) * Kdim + k0 + c * 2) = v;
    }
  }
}

// One wave (32 lanes) per token: logits, softmax, top-2, scatter to expert lists.
__global__ __launch_bounds__(256) void moe_router(
    const float* __restrict__ x, const float* __restrict__ gw,
    float* __restrict__ logits_out, int* __restrict__ cnt,
    int* __restrict__ tok, float* __restrict__ wslot)
{
  const int lane = threadIdx.x & 31;
  const int wid  = threadIdx.x >> 5;
  const int t    = blockIdx.x * 8 + wid;

  float acc[E_NUM];
#pragma unroll
  for (int e = 0; e < E_NUM; ++e) acc[e] = 0.f;

  const float* xr = x + (size_t)t * H_DIM;
  for (int h = lane; h < H_DIM; h += 32) {
    float xv = xr[h];
    const float* g = gw + h * E_NUM;
#pragma unroll
    for (int e = 0; e < E_NUM; ++e) acc[e] += xv * g[e];
  }
#pragma unroll
  for (int e = 0; e < E_NUM; ++e) {
#pragma unroll
    for (int off = 16; off > 0; off >>= 1)
      acc[e] += __shfl_xor(acc[e], off, 32);
  }

  if (lane == 0) {
    float m = acc[0];
#pragma unroll
    for (int e = 1; e < E_NUM; ++e) m = fmaxf(m, acc[e]);
    float p[E_NUM], sum = 0.f;
#pragma unroll
    for (int e = 0; e < E_NUM; ++e) { p[e] = __expf(acc[e] - m); sum += p[e]; }
    float inv = 1.0f / sum;

    int e0 = 0; float b0 = -1.f;
#pragma unroll
    for (int e = 0; e < E_NUM; ++e) { float pe = p[e] * inv; if (pe > b0) { b0 = pe; e0 = e; } }
    int e1 = 0; float b1 = -1.f;
#pragma unroll
    for (int e = 0; e < E_NUM; ++e) {
      if (e == e0) continue;
      float pe = p[e] * inv; if (pe > b1) { b1 = pe; e1 = e; }
    }
#pragma unroll
    for (int e = 0; e < E_NUM; ++e) logits_out[t * E_NUM + e] = acc[e];

    wslot[t * K_TOP + 0] = b0;
    wslot[t * K_TOP + 1] = b1;
    int p0 = atomicAdd(&cnt[e0], 1);
    tok[e0 * T_TOK + p0] = t * K_TOP + 0;
    int p1 = atomicAdd(&cnt[e1], 1);
    tok[e1 * T_TOK + p1] = t * K_TOP + 1;
  }
}

// Grouped GEMM 1: mid = silu(X@Wg) * (X@Wu). bf16 operands, B pre-transposed to
// [N][K]. Block tile M=64 x N=128, K-step 32, 8 waves x (32x32). Staging uses
// double-buffered async global->LDS copies (ASYNCcnt), written as an explicit
// two-phase loop so each phase sees compile-time LDS buffer addresses (keeps
// accumulators pinned; no buffer-select moves in the WMMA stream).
__global__ __launch_bounds__(256) void moe_gemm_gateup(
    const bf16_t* __restrict__ xbf, const bf16_t* __restrict__ WgT,
    const bf16_t* __restrict__ WuT, const int* __restrict__ cnt,
    const int* __restrict__ tok, bf16_t* __restrict__ mid)
{
  const int e  = blockIdx.z;
  const int n0 = blockIdx.x * 128;
  const int m0 = blockIdx.y * 64;
  const int ce = cnt[e];
  if (m0 >= ce) return;

  __shared__ bf16_t As[2][64 * 40];
  __shared__ bf16_t Bg[2][128 * 40];
  __shared__ bf16_t Bu[2][128 * 40];
  __shared__ int Sl[64];

  const int tid  = threadIdx.x;
  const int lane = tid & 31;
  const int wid  = tid >> 5;
  const int wm   = wid >> 2;   // 0..1
  const int wn   = wid & 3;    // 0..3

  if (tid < 64) {
    int idx = m0 + tid;
    Sl[tid] = (idx < ce) ? tok[e * T_TOK + idx] : -1;
  }
  __syncthreads();

  // ---- k-invariant staging assignments ----
  const int ar = tid >> 2;
  const int ac = tid & 3;
  const int aslot = Sl[ar];
  const bf16_t* asrc =
      xbf + (size_t)((aslot < 0 ? 0 : aslot) >> 1) * H_DIM + ac * 8;
  const unsigned aoff0 = (unsigned)(size_t)&As[0][ar * 40 + ac * 8];
  const unsigned aoff1 = (unsigned)(size_t)&As[1][ar * 40 + ac * 8];

  const bf16_t* wg_base = WgT + (size_t)e * H_DIM * I_DIM;   // [I][H]
  const bf16_t* wu_base = WuT + (size_t)e * H_DIM * I_DIM;
  const bf16_t* gsrc[2];
  const bf16_t* usrc[2];
  unsigned goff[2][2], uoff[2][2];                            // [buf][i]
#pragma unroll
  for (int i = 0; i < 2; ++i) {
    int eid = tid + i * 256;
    int c = eid & 3, n = eid >> 2;
    gsrc[i] = wg_base + (size_t)(n0 + n) * H_DIM + c * 8;
    usrc[i] = wu_base + (size_t)(n0 + n) * H_DIM + c * 8;
    goff[0][i] = (unsigned)(size_t)&Bg[0][n * 40 + c * 8];
    goff[1][i] = (unsigned)(size_t)&Bg[1][n * 40 + c * 8];
    uoff[0][i] = (unsigned)(size_t)&Bu[0][n * 40 + c * 8];
    uoff[1][i] = (unsigned)(size_t)&Bu[1][n * 40 + c * 8];
  }

  v8f zero8 = {};
  v8f cg[2][2], cu[2][2];
#pragma unroll
  for (int i = 0; i < 2; ++i)
#pragma unroll
    for (int j = 0; j < 2; ++j) { cg[i][j] = zero8; cu[i][j] = zero8; }

  auto stage0 = [&](int h0) {   // fill buffer 0
    async_load_b128(aoff0, asrc + h0);
    async_load_b128(goff[0][0], gsrc[0] + h0);
    async_load_b128(goff[0][1], gsrc[1] + h0);
    async_load_b128(uoff[0][0], usrc[0] + h0);
    async_load_b128(uoff[0][1], usrc[1] + h0);
  };
  auto stage1 = [&](int h0) {   // fill buffer 1
    async_load_b128(aoff1, asrc + h0);
    async_load_b128(goff[1][0], gsrc[0] + h0);
    async_load_b128(goff[1][1], gsrc[1] + h0);
    async_load_b128(uoff[1][0], usrc[0] + h0);
    async_load_b128(uoff[1][1], usrc[1] + h0);
  };
  auto compute = [&](const bf16_t* Ab, const bf16_t* Gb, const bf16_t* Ub) {
#pragma unroll
    for (int tm = 0; tm < 2; ++tm) {
      v16bf a = load_frag(Ab + (wm * 32 + tm * 16) * 40, 40, lane);
#pragma unroll
      for (int tn = 0; tn < 2; ++tn) {
        v16bf bg = load_frag(Gb + (wn * 32 + tn * 16) * 40, 40, lane);
        v16bf bu = load_frag(Ub + (wn * 32 + tn * 16) * 40, 40, lane);
        cg[tm][tn] = __builtin_amdgcn_wmma_f32_16x16x32_bf16(
            false, a, false, bg, (short)0, cg[tm][tn], false, false);
        cu[tm][tn] = __builtin_amdgcn_wmma_f32_16x16x32_bf16(
            false, a, false, bu, (short)0, cu[tm][tn], false, false);
      }
    }
  };

  const int nIter = H_DIM / 32;      // 32 iterations: even
  stage0(0);
  for (int it = 0; it < nIter; it += 2) {
    // ---- phase A: consume buffer 0, prefetch buffer 1 ----
    __syncthreads();                 // buffer 1 free
    stage1((it + 1) * 32);           // it+1 < nIter (nIter even)
    wait_asynccnt<5>();              // buffer-0 loads landed (in-order)
    __syncthreads();                 // all waves' buffer-0 data visible
    compute(&As[0][0], &Bg[0][0], &Bu[0][0]);
    // ---- phase B: consume buffer 1, prefetch buffer 0 ----
    __syncthreads();                 // buffer 0 free
    if (it + 2 < nIter) {
      stage0((it + 2) * 32);
      wait_asynccnt<5>();
    } else {
      wait_asynccnt<0>();
    }
    __syncthreads();
    compute(&As[1][0], &Bg[1][0], &Bu[1][0]);
  }

  // Epilogue: mid = silu(g)*u -> bf16, scattered by slot.
  const int rsel = (lane >> 4) * 8;
  const int col  = lane & 15;
#pragma unroll
  for (int tm = 0; tm < 2; ++tm)
#pragma unroll
    for (int tn = 0; tn < 2; ++tn)
#pragma unroll
      for (int r = 0; r < 8; ++r) {
        int mloc = wm * 32 + tm * 16 + r + rsel;
        int slot = Sl[mloc];
        if (slot < 0) continue;
        int n = n0 + wn * 32 + tn * 16 + col;
        float g = cg[tm][tn][r], u = cu[tm][tn][r];
        float s = g / (1.0f + __expf(-g));
        mid[(size_t)slot * I_DIM + n] = f2bf(s * u);
      }
}

// Grouped GEMM 2: out_slot = (mid @ Wd) * route_weight, WdT pre-transposed [H][I].
__global__ __launch_bounds__(256) void moe_gemm_down(
    const bf16_t* __restrict__ mid, const bf16_t* __restrict__ WdT,
    const int* __restrict__ cnt, const int* __restrict__ tok,
    const float* __restrict__ wslot, float* __restrict__ oslot)
{
  const int e  = blockIdx.z;
  const int n0 = blockIdx.x * 128;   // over H
  const int m0 = blockIdx.y * 64;
  const int ce = cnt[e];
  if (m0 >= ce) return;

  __shared__ bf16_t As[2][64 * 40];
  __shared__ bf16_t Bs[2][128 * 40];
  __shared__ int Sl[64];

  const int tid  = threadIdx.x;
  const int lane = tid & 31;
  const int wid  = tid >> 5;
  const int wm   = wid >> 2;
  const int wn   = wid & 3;

  if (tid < 64) {
    int idx = m0 + tid;
    Sl[tid] = (idx < ce) ? tok[e * T_TOK + idx] : -1;
  }
  __syncthreads();

  const int ar = tid >> 2;
  const int ac = tid & 3;
  const int aslot = Sl[ar];
  const bf16_t* asrc =
      mid + (size_t)(aslot < 0 ? 0 : aslot) * I_DIM + ac * 8;
  const unsigned aoff0 = (unsigned)(size_t)&As[0][ar * 40 + ac * 8];
  const unsigned aoff1 = (unsigned)(size_t)&As[1][ar * 40 + ac * 8];

  const bf16_t* wd_base = WdT + (size_t)e * I_DIM * H_DIM;   // [H][I]
  const bf16_t* bsrc[2];
  unsigned boff[2][2];                                       // [buf][i]
#pragma unroll
  for (int i = 0; i < 2; ++i) {
    int eid = tid + i * 256;
    int c = eid & 3, n = eid >> 2;
    bsrc[i] = wd_base + (size_t)(n0 + n) * I_DIM + c * 8;
    boff[0][i] = (unsigned)(size_t)&Bs[0][n * 40 + c * 8];
    boff[1][i] = (unsigned)(size_t)&Bs[1][n * 40 + c * 8];
  }

  v8f zero8 = {};
  v8f c[2][2];
#pragma unroll
  for (int i = 0; i < 2; ++i)
#pragma unroll
    for (int j = 0; j < 2; ++j) c[i][j] = zero8;

  auto stage0 = [&](int k0) {
    async_load_b128(aoff0, asrc + k0);
    async_load_b128(boff[0][0], bsrc[0] + k0);
    async_load_b128(boff[0][1], bsrc[1] + k0);
  };
  auto stage1 = [&](int k0) {
    async_load_b128(aoff1, asrc + k0);
    async_load_b128(boff[1][0], bsrc[0] + k0);
    async_load_b128(boff[1][1], bsrc[1] + k0);
  };
  auto compute = [&](const bf16_t* Ab, const bf16_t* Bb) {
#pragma unroll
    for (int tm = 0; tm < 2; ++tm) {
      v16bf a = load_frag(Ab + (wm * 32 + tm * 16) * 40, 40, lane);
#pragma unroll
      for (int tn = 0; tn < 2; ++tn) {
        v16bf bfrag = load_frag(Bb + (wn * 32 + tn * 16) * 40, 40, lane);
        c[tm][tn] = __builtin_amdgcn_wmma_f32_16x16x32_bf16(
            false, a, false, bfrag, (short)0, c[tm][tn], false, false);
      }
    }
  };

  const int nIter = I_DIM / 32;      // 64 iterations: even
  stage0(0);
  for (int it = 0; it < nIter; it += 2) {
    __syncthreads();
    stage1((it + 1) * 32);
    wait_asynccnt<3>();
    __syncthreads();
    compute(&As[0][0], &Bs[0][0]);

    __syncthreads();
    if (it + 2 < nIter) {
      stage0((it + 2) * 32);
      wait_asynccnt<3>();
    } else {
      wait_asynccnt<0>();
    }
    __syncthreads();
    compute(&As[1][0], &Bs[1][0]);
  }

  const int rsel = (lane >> 4) * 8;
  const int col  = lane & 15;
#pragma unroll
  for (int tm = 0; tm < 2; ++tm)
#pragma unroll
    for (int tn = 0; tn < 2; ++tn)
#pragma unroll
      for (int r = 0; r < 8; ++r) {
        int mloc = wm * 32 + tm * 16 + r + rsel;
        int slot = Sl[mloc];
        if (slot < 0) continue;
        float w = wslot[slot];
        int n = n0 + wn * 32 + tn * 16 + col;
        oslot[(size_t)slot * H_DIM + n] = c[tm][tn][r] * w;
      }
}

// final[t][h] = out_slot[t*2][h] + out_slot[t*2+1][h]
__global__ __launch_bounds__(256) void moe_combine(const float* __restrict__ os,
                                                   float* __restrict__ out) {
  int i  = blockIdx.x * 256 + threadIdx.x;   // over T*H/4
  int t  = i / (H_DIM / 4);
  int h4 = i % (H_DIM / 4);
  float4 a = ((const float4*)(os + (size_t)(2 * t)     * H_DIM))[h4];
  float4 b = ((const float4*)(os + (size_t)(2 * t + 1) * H_DIM))[h4];
  float4 r; r.x = a.x + b.x; r.y = a.y + b.y; r.z = a.z + b.z; r.w = a.w + b.w;
  ((float4*)(out + (size_t)t * H_DIM))[h4] = r;
}

// ---------------- workspace layout ----------------
static constexpr size_t OFF_CNT   = 0;                                            // E ints
static constexpr size_t OFF_TOK   = 256;                                          // E*T ints
static constexpr size_t OFF_WSLOT = OFF_TOK + (size_t)E_NUM * T_TOK * 4;          // T*K f32
static constexpr size_t OFF_XBF   = OFF_WSLOT + (size_t)T_TOK * K_TOP * 4;        // T*H bf16
static constexpr size_t OFF_MID   = OFF_XBF + (size_t)T_TOK * H_DIM * 2;          // T*K*I bf16
static constexpr size_t OFF_OSLOT = OFF_MID + (size_t)T_TOK * K_TOP * I_DIM * 2;  // T*K*H f32
static constexpr size_t OFF_WGT   = OFF_OSLOT + (size_t)T_TOK * K_TOP * H_DIM * 4;
static constexpr size_t OFF_WUT   = OFF_WGT + (size_t)E_NUM * H_DIM * I_DIM * 2;
static constexpr size_t OFF_WDT   = OFF_WUT + (size_t)E_NUM * H_DIM * I_DIM * 2;

extern "C" void kernel_launch(void* const* d_in, const int* in_sizes, int n_in,
                              void* d_out, int out_size, void* d_ws, size_t ws_size,
                              hipStream_t stream) {
  const float* x  = (const float*)d_in[0];   // [B,S,H]
  const float* gw = (const float*)d_in[1];   // [H,E]
  const float* Wg = (const float*)d_in[2];   // [E,H,I]
  const float* Wu = (const float*)d_in[3];   // [E,H,I]
  const float* Wd = (const float*)d_in[4];   // [E,I,H]
  (void)in_sizes; (void)n_in; (void)out_size; (void)ws_size;

  float* out    = (float*)d_out;                       // final [T,H]
  float* logits = out + (size_t)T_TOK * H_DIM;         // router_logits [T,E]

  char*   ws    = (char*)d_ws;
  int*    cnt   = (int*)(ws + OFF_CNT);
  int*    tok   = (int*)(ws + OFF_TOK);
  float*  wslot = (float*)(ws + OFF_WSLOT);
  bf16_t* xbf   = (bf16_t*)(ws + OFF_XBF);
  bf16_t* mid   = (bf16_t*)(ws + OFF_MID);
  float*  oslot = (float*)(ws + OFF_OSLOT);
  bf16_t* WgT   = (bf16_t*)(ws + OFF_WGT);
  bf16_t* WuT   = (bf16_t*)(ws + OFF_WUT);
  bf16_t* WdT   = (bf16_t*)(ws + OFF_WDT);

  moe_init<<<1, 64, 0, stream>>>(cnt);
  moe_cvt_x<<<(T_TOK * H_DIM / 4) / 256, 256, 0, stream>>>((const float4*)x, (uint2*)xbf);
  // Weight transpose+convert: [K][N] fp32 -> [N][K] bf16 per expert.
  moe_transpose_cvt<<<dim3(H_DIM / 64, I_DIM / 32, E_NUM), 256, 0, stream>>>(Wg, WgT, H_DIM, I_DIM);
  moe_transpose_cvt<<<dim3(H_DIM / 64, I_DIM / 32, E_NUM), 256, 0, stream>>>(Wu, WuT, H_DIM, I_DIM);
  moe_transpose_cvt<<<dim3(I_DIM / 64, H_DIM / 32, E_NUM), 256, 0, stream>>>(Wd, WdT, I_DIM, H_DIM);
  moe_router<<<T_TOK / 8, 256, 0, stream>>>(x, gw, logits, cnt, tok, wslot);
  moe_gemm_gateup<<<dim3(I_DIM / 128, T_TOK / 64, E_NUM), 256, 0, stream>>>(
      xbf, WgT, WuT, cnt, tok, mid);
  moe_gemm_down<<<dim3(H_DIM / 128, T_TOK / 64, E_NUM), 256, 0, stream>>>(
      mid, WdT, cnt, tok, wslot, oslot);
  moe_combine<<<(T_TOK * H_DIM / 4) / 256, 256, 0, stream>>>(oslot, out);
}
